// BestRQModel_33818572489289
// MI455X (gfx1250) — matrix operand order repaired and checked by reading, not compiled, not run on hardware
//
#include <hip/hip_runtime.h>
#include <hip/hip_bf16.h>
#include <stdint.h>

typedef __attribute__((ext_vector_type(16))) __bf16 v16bf;
typedef __attribute__((ext_vector_type(8)))  __bf16 bf16x8;
typedef __attribute__((ext_vector_type(8)))  float  v8f;

#define B_    8
#define T_    1024
#define F_    80
#define TS_   256
#define G_    4
#define E_    16
#define GE_   64
#define C_    8192
#define D_    512
#define SF_   320
#define NROW  2048            // B_*TS_
#define INFF  __builtin_inff()

// ws layout (bytes)
#define WS_ACCUM   0          // 4 floats
#define WS_BITMAP  64         // 256 u32
#define WS_MROW    2048       // 2048 f32
#define WS_TARGET  10240      // 2048*4 i32
#define WS_LAT     43008      // 2048*64 f32
#define WS_CN      567296     // 4*8192 f32 (column norms)
#define WS_PB      698368     // packed bf16 weights: 4*512*8192*2 = 32MB
#define WS_PB_SZ   33554432

// ---------------- helpers ----------------

__device__ __forceinline__ float wave_sum32(float v) {
#pragma unroll
  for (int d = 1; d < 32; d <<= 1) v += __shfl_xor(v, d, 32);
  return v;
}

// A-fragment for v_wmma_f32_16x16x32_bf16 from an LDS row (row-major K).
// contiguous K order (B uses the same): lane<16 -> K = kb+[0..15];
//                  lane>=16 -> K = kb+16+[0..15].
__device__ __forceinline__ v16bf load_afrag(const __bf16* rowp, int kb, int half) {
  bf16x8 lo = *(const bf16x8*)(rowp + kb + half * 16);
  bf16x8 hi = *(const bf16x8*)(rowp + kb + half * 16 + 8);
  v16bf a;
#pragma unroll
  for (int i = 0; i < 8; ++i) { a[i] = lo[i]; a[i + 8] = hi[i]; }
  return a;
}

__device__ __forceinline__ v16bf join8(bf16x8 lo, bf16x8 hi) {
  v16bf a;
#pragma unroll
  for (int i = 0; i < 8; ++i) { a[i] = lo[i]; a[i + 8] = hi[i]; }
  return a;
}

// ---------------- kernel 1a: features_pen ----------------

__global__ __launch_bounds__(256) void k_featpen(const float* __restrict__ feats,
                                                 const float* __restrict__ mean,
                                                 const float* __restrict__ istd,
                                                 float* __restrict__ accum) {
  float s = 0.f;
  for (int i = blockIdx.x * 256 + threadIdx.x; i < B_ * T_ * F_; i += gridDim.x * 256) {
    int f = i % F_;
    float v = (feats[i] - mean[f]) * istd[f];
    s += v * v;
  }
  __shared__ float red[8];
  int lane = threadIdx.x & 31, wv = threadIdx.x >> 5;
  s = wave_sum32(s);
  if (lane == 0) red[wv] = s;
  __syncthreads();
  if (threadIdx.x == 0) {
    float t = 0.f;
#pragma unroll
    for (int w = 0; w < 8; ++w) t += red[w];
    atomicAdd(&accum[0], t);
  }
}

// ---------------- kernel 1b: stack + LayerNorm + projection + mask ----------------

__global__ __launch_bounds__(256) void k_prep(const float* __restrict__ feats,
                                              const int* __restrict__ lens,
                                              const unsigned char* __restrict__ masks,
                                              const float* __restrict__ mean,
                                              const float* __restrict__ istd,
                                              const float* __restrict__ P,   // [320,64]
                                              float* __restrict__ lat,       // [2048,64]
                                              float* __restrict__ mrow,      // [2048]
                                              float* __restrict__ accum) {
  __shared__ float sN[8][SF_];
  int lane = threadIdx.x & 31, wv = threadIdx.x >> 5;
  int row = blockIdx.x * 8 + wv;          // 256 blocks x 8 waves = 2048 rows
  int b = row >> 8, t = row & 255;

  float x[10];
  float sum = 0.f, sq = 0.f;
#pragma unroll
  for (int k = 0; k < 10; ++k) {
    int j = lane + 32 * k;                 // 0..319
    int tt = t * 4 + (j / F_), f = j % F_;
    float v = (feats[(b * T_ + tt) * F_ + f] - mean[f]) * istd[f];
    x[k] = v; sum += v; sq += v * v;
  }
  sum = wave_sum32(sum);
  sq = wave_sum32(sq);
  float mu = sum * (1.f / SF_);
  float var = sq * (1.f / SF_) - mu * mu;
  float rs = rsqrtf(var + 1e-5f);
#pragma unroll
  for (int k = 0; k < 10; ++k) sN[wv][lane + 32 * k] = (x[k] - mu) * rs;
  __syncthreads();

  for (int o = lane; o < GE_; o += 32) {
    float acc = 0.f;
    for (int j = 0; j < SF_; ++j) acc += sN[wv][j] * P[j * GE_ + o];
    lat[row * GE_ + o] = acc;
  }
  if (lane == 0) {
    const unsigned char* mp = masks + b * T_ + t * 4;
    bool cm = mp[0] && mp[1] && mp[2] && mp[3];
    int ls = (lens[b] - 4) / 4 + 1; if (ls < 0) ls = 0;
    float mval = (t < ls && cm) ? 1.f : 0.f;
    mrow[row] = mval;
    atomicAdd(&accum[1], mval);
  }
}

// ---------------- column norms of the codebook ----------------

__global__ __launch_bounds__(256) void k_norms(const float* __restrict__ emb,  // [C,G,E]
                                               float* __restrict__ cn) {       // [G,C]
  int idx = blockIdx.x * 256 + threadIdx.x;   // g*C + c
  int c = idx & (C_ - 1);
  int g = idx >> 13;
  const float* ep = emb + ((size_t)c * G_ + g) * E_;
  float s = 0.f;
#pragma unroll
  for (int e = 0; e < 16; ++e) { float xx = ep[e]; s += xx * xx; }
  cn[idx] = s;
}

// ---------------- weight repack: W[g][k][c] f32 -> PB fragment-native bf16 ----------------
// PB[((g*16 + kb/32)*C + c)*32 + half*16 + e] = (bf16) W[g][kb + half*16 + e][c]
// B-fragment for a lane (half, col c) is then 32 contiguous bytes -> 2x global_load_b128.

__global__ __launch_bounds__(256) void k_pack(const float* __restrict__ W,
                                              __bf16* __restrict__ PB) {
  int idx = blockIdx.x * 256 + threadIdx.x;   // (g, kblk, half, c)
  int c = idx & (C_ - 1);
  int rest = idx >> 13;
  int half = rest & 1;
  int kblk = (rest >> 1) & 15;
  int g = rest >> 5;
  const float* src = W + ((size_t)g * D_ + kblk * 32 + half * 16) * C_ + c;
  __bf16* dst = PB + (((size_t)g * 16 + kblk) * C_ + c) * 32 + half * 16;
#pragma unroll
  for (int e = 0; e < 16; ++e) dst[e] = (__bf16)src[(size_t)e * C_];
}

// ---------------- kernel 2: quantize codes via WMMA ----------------
// per block: one g, 32 rows of lat; argmin_c  ||emb_c||^2 - 2 lat.emb_c

__global__ __launch_bounds__(256) void k_quant(const float* __restrict__ lat,  // [2048,64]
                                               const float* __restrict__ emb,  // [C,G,E]
                                               const float* __restrict__ cn,   // [G,C]
                                               int* __restrict__ target) {     // [2048,G]
  __shared__ __bf16 sA[32][40];            // K padded 16->32, stride 40 (b128-friendly)
  __shared__ float sMn[8][32];
  __shared__ int   sMi[8][32];
  int mbase = blockIdx.x * 32;
  int g = blockIdx.y;

  for (int i = threadIdx.x; i < 32 * 32; i += 256) {
    int r = i >> 5, k = i & 31;
    sA[r][k] = (k < E_) ? (__bf16)lat[(mbase + r) * GE_ + g * E_ + k] : (__bf16)0.f;
  }
  __syncthreads();

  int lane = threadIdx.x & 31, wv = threadIdx.x >> 5;
  int half = lane >> 4, nloc = lane & 15;

  float mn[16]; int mi[16];
#pragma unroll
  for (int s = 0; s < 16; ++s) { mn[s] = INFF; mi[s] = 0; }

  v16bf a0 = load_afrag(&sA[nloc][0], 0, half);
  v16bf a1 = load_afrag(&sA[nloc + 16][0], 0, half);
  const float* cng = cn + (g << 13);

  for (int cb = wv * 16; cb < C_; cb += 128) {
    int c = cb + nloc;
    v16bf bfrag;
    if (half == 0) {                        // real K rows 0..15 live in lanes 0-15
      const float* ep = emb + ((size_t)c * G_ + g) * E_;
#pragma unroll
      for (int e = 0; e < 16; ++e) bfrag[e] = (__bf16)ep[e];
    } else {                                // K rows 16..31 are the zero pad
#pragma unroll
      for (int e = 0; e < 16; ++e) bfrag[e] = (__bf16)0.f;
    }
    float nrm = cng[c];                     // no cross-lane traffic in the hot loop

    v8f z = {0.f, 0.f, 0.f, 0.f, 0.f, 0.f, 0.f, 0.f};
    v8f d0 = __builtin_amdgcn_wmma_f32_16x16x32_bf16(false, a0, false, bfrag, (short)0, z, false, false);
    v8f d1 = __builtin_amdgcn_wmma_f32_16x16x32_bf16(false, a1, false, bfrag, (short)0, z, false, false);
#pragma unroll
    for (int j = 0; j < 8; ++j) {
      float q0 = nrm - 2.f * d0[j];
      if (q0 < mn[j]) { mn[j] = q0; mi[j] = c; }
      float q1 = nrm - 2.f * d1[j];
      if (q1 < mn[j + 8]) { mn[j + 8] = q1; mi[j + 8] = c; }
    }
  }

#pragma unroll
  for (int d = 1; d < 16; d <<= 1) {
#pragma unroll
    for (int s = 0; s < 16; ++s) {
      float om = __shfl_xor(mn[s], d, 32);
      int   oi = __shfl_xor(mi[s], d, 32);
      if (om < mn[s] || (om == mn[s] && oi < mi[s])) { mn[s] = om; mi[s] = oi; }
    }
  }
  if (nloc == 0) {
#pragma unroll
    for (int j = 0; j < 8; ++j) {
      int r0 = j + 8 * half, r1 = 16 + j + 8 * half;
      sMn[wv][r0] = mn[j];     sMi[wv][r0] = mi[j];
      sMn[wv][r1] = mn[j + 8]; sMi[wv][r1] = mi[j + 8];
    }
  }
  __syncthreads();
  if (threadIdx.x < 32) {
    int r = threadIdx.x;
    float bm = INFF; int bi = 0;
#pragma unroll
    for (int w = 0; w < 8; ++w) {
      float m = sMn[w][r]; int i = sMi[w][r];
      if (m < bm || (m == bm && i < bi)) { bm = m; bi = i; }
    }
    target[(mbase + r) * G_ + g] = bi;
  }
}

// ---------------- kernel 3: logits WMMA + fused online softmax / argmax / gather ----------------

template <bool PACKED>
__global__ __launch_bounds__(256) void k_logits_t(const float* __restrict__ enc,   // [2048,512]
                                                  const float* __restrict__ W,     // [G,512,C]
                                                  const __bf16* __restrict__ PB,   // packed weights
                                                  const float* __restrict__ bias,  // [G,C]
                                                  const int* __restrict__ target,  // [2048,G]
                                                  const float* __restrict__ mrow,  // [2048]
                                                  float* __restrict__ accum,       // [4]
                                                  unsigned* __restrict__ bitmap) { // [C/32]
  __shared__ __bf16 sA[32][520];           // 32 rows x 512 K, stride 520 (conflict-free b128)
  __shared__ float sMax[8][32], sSum[8][32], sAmv[8][32], sTlg[8][32];
  __shared__ int   sAmi[8][32];
  int mbase = blockIdx.x * 32;
  int g = blockIdx.y;

  for (int i = threadIdx.x; i < 32 * D_; i += 256) {
    int r = i >> 9, k = i & 511;
    sA[r][k] = (__bf16)enc[(size_t)(mbase + r) * D_ + k];
  }
  __syncthreads();

  int lane = threadIdx.x & 31, wv = threadIdx.x >> 5;
  int half = lane >> 4, nloc = lane & 15;

  float mx[16], se[16], av[16], tl[16];
  int ai[16], tg[16];
#pragma unroll
  for (int s = 0; s < 16; ++s) { mx[s] = -INFF; se[s] = 0.f; av[s] = -INFF; ai[s] = 0; tl[s] = -INFF; }
#pragma unroll
  for (int j = 0; j < 8; ++j) {
    tg[j]     = target[(mbase + j + 8 * half) * G_ + g];
    tg[j + 8] = target[(mbase + 16 + j + 8 * half) * G_ + g];
  }

  const float*  Wg  = W + (size_t)g * D_ * C_;
  const __bf16* PBg = PB + (size_t)g * 16 * C_ * 32;

  for (int cb = wv * 16; cb < C_; cb += 128) {
    const int c = cb + nloc;
    v8f acc0 = {0.f, 0.f, 0.f, 0.f, 0.f, 0.f, 0.f, 0.f};
    v8f acc1 = acc0;
    const float*  wp = Wg + (size_t)(half * 16) * C_ + c;
    const __bf16* pb = PBg + (size_t)c * 32 + half * 16;
    for (int kb = 0; kb < D_; kb += 32) {
      v16bf bfrag;
      if (PACKED) {
        const __bf16* pk = pb + (size_t)(kb >> 5) * (C_ * 32);
        bf16x8 b0 = *(const bf16x8*)pk;
        bf16x8 b1 = *(const bf16x8*)(pk + 8);
        bfrag = join8(b0, b1);
        // pull next slab's chunk for this k-block toward the WGP while WMMAs run
        __builtin_prefetch(pk + 128 * 32, 0, 3);
      } else {
        const float* wk = wp + (size_t)kb * C_;
#pragma unroll
        for (int e = 0; e < 16; ++e) bfrag[e] = (__bf16)wk[(size_t)e * C_];
      }
      v16bf a0 = load_afrag(&sA[nloc][0], kb, half);
      v16bf a1 = load_afrag(&sA[nloc + 16][0], kb, half);
      acc0 = __builtin_amdgcn_wmma_f32_16x16x32_bf16(false, a0, false, bfrag, (short)0, acc0, false, false);
      acc1 = __builtin_amdgcn_wmma_f32_16x16x32_bf16(false, a1, false, bfrag, (short)0, acc1, false, false);
    }
    float bs = bias[g * C_ + c];
#pragma unroll
    for (int j = 0; j < 8; ++j) {
      {
        float v = acc0[j] + bs; int s = j;
        if (v > mx[s]) { se[s] = se[s] * __expf(mx[s] - v) + 1.f; mx[s] = v; }
        else           { se[s] += __expf(v - mx[s]); }
        if (v > av[s]) { av[s] = v; ai[s] = c; }
        if (c == tg[s]) tl[s] = v;
      }
      {
        float v = acc1[j] + bs; int s = j + 8;
        if (v > mx[s]) { se[s] = se[s] * __expf(mx[s] - v) + 1.f; mx[s] = v; }
        else           { se[s] += __expf(v - mx[s]); }
        if (v > av[s]) { av[s] = v; ai[s] = c; }
        if (c == tg[s]) tl[s] = v;
      }
    }
  }

  // reduce across the 16 lanes of each half-wave
#pragma unroll
  for (int d = 1; d < 16; d <<= 1) {
#pragma unroll
    for (int s = 0; s < 16; ++s) {
      float om = __shfl_xor(mx[s], d, 32);
      float os = __shfl_xor(se[s], d, 32);
      float nm = fmaxf(mx[s], om);
      se[s] = se[s] * __expf(mx[s] - nm) + os * __expf(om - nm);
      mx[s] = nm;
      float oav = __shfl_xor(av[s], d, 32);
      int   oai = __shfl_xor(ai[s], d, 32);
      if (oav > av[s] || (oav == av[s] && oai < ai[s])) { av[s] = oav; ai[s] = oai; }
      float otl = __shfl_xor(tl[s], d, 32);
      tl[s] = fmaxf(tl[s], otl);
    }
  }
  if (nloc == 0) {
#pragma unroll
    for (int j = 0; j < 8; ++j) {
      int r0 = j + 8 * half, r1 = 16 + j + 8 * half;
      sMax[wv][r0] = mx[j];     sSum[wv][r0] = se[j];
      sAmv[wv][r0] = av[j];     sAmi[wv][r0] = ai[j];   sTlg[wv][r0] = tl[j];
      sMax[wv][r1] = mx[j + 8]; sSum[wv][r1] = se[j + 8];
      sAmv[wv][r1] = av[j + 8]; sAmi[wv][r1] = ai[j + 8]; sTlg[wv][r1] = tl[j + 8];
    }
  }
  __syncthreads();

  if (threadIdx.x < 32) {
    int r = threadIdx.x;
    float M = -INFF, S = 0.f, AV = -INFF, TL = -INFF;
    int AI = 0x7fffffff;
#pragma unroll
    for (int w = 0; w < 8; ++w) {
      float mw = sMax[w][r], sw = sSum[w][r];
      float nM = fmaxf(M, mw);
      S = S * __expf(M - nM) + sw * __expf(mw - nM);
      M = nM;
      float avw = sAmv[w][r]; int aiw = sAmi[w][r];
      if (avw > AV || (avw == AV && aiw < AI)) { AV = avw; AI = aiw; }
      TL = fmaxf(TL, sTlg[w][r]);
    }
    int row = mbase + r;
    float mval = mrow[row];
    float per = -(TL - M - __logf(S));
    int tcode = target[row * G_ + g];
    float corr = (AI == tcode) ? 1.f : 0.f;
    atomicAdd(&accum[2], per * mval);
    atomicAdd(&accum[3], corr * mval);
    int val = (mval > 0.f) ? tcode : 0;
    atomicOr(&bitmap[val >> 5], 1u << (val & 31));
  }
}

// ---------------- kernel 4: finalize ----------------

__global__ __launch_bounds__(256) void k_final(const float* __restrict__ accum,
                                               const unsigned* __restrict__ bitmap,
                                               float* __restrict__ out) {
  __shared__ int cnt[8];
  int lane = threadIdx.x & 31, wv = threadIdx.x >> 5;
  int c = __popc(bitmap[threadIdx.x]);     // C/32 = 256 words, one per thread
#pragma unroll
  for (int d = 1; d < 32; d <<= 1) c += __shfl_xor(c, d, 32);
  if (lane == 0) cnt[wv] = c;
  __syncthreads();
  if (threadIdx.x == 0) {
    int tot = 0;
#pragma unroll
    for (int w = 0; w < 8; ++w) tot += cnt[w];
    float featpen = accum[0] * (1.f / (B_ * T_ * F_));
    float msum = accum[1];
    float denom = msum + 1e-5f;
    float loss = accum[2] / (denom * (float)G_) + 0.01f * featpen;
    float num_codes = msum * (float)G_;
    float cacc = accum[3] / num_codes;
    out[0] = loss;
    out[1] = cacc;
    out[2] = featpen;
    out[3] = num_codes;
    out[4] = (float)tot;
  }
}

// ---------------- launch ----------------

extern "C" void kernel_launch(void* const* d_in, const int* in_sizes, int n_in,
                              void* d_out, int out_size, void* d_ws, size_t ws_size,
                              hipStream_t stream) {
  const float*         feats = (const float*)d_in[0];
  const int*           lens  = (const int*)d_in[1];
  const unsigned char* masks = (const unsigned char*)d_in[2];   // bool [B,T]
  /* d_in[3] mask_emb unused (masked_xs is not an output) */
  const float* smean = (const float*)d_in[4];
  const float* sistd = (const float*)d_in[5];
  const float* P     = (const float*)d_in[6];
  const float* emb   = (const float*)d_in[7];
  const float* W     = (const float*)d_in[8];
  const float* bias  = (const float*)d_in[9];
  const float* enc   = (const float*)d_in[10];
  float* out = (float*)d_out;

  char* ws = (char*)d_ws;
  float*    accum  = (float*)(ws + WS_ACCUM);
  unsigned* bitmap = (unsigned*)(ws + WS_BITMAP);
  float*    mrow   = (float*)(ws + WS_MROW);
  int*      target = (int*)(ws + WS_TARGET);
  float*    lat    = (float*)(ws + WS_LAT);
  float*    cn     = (float*)(ws + WS_CN);
  __bf16*   PB     = (__bf16*)(ws + WS_PB);

  const bool packed = ws_size >= (size_t)WS_PB + (size_t)WS_PB_SZ;  // fixed per run

  hipMemsetAsync(ws, 0, 2048, stream);              // zero accumulators + bitmap

  k_featpen<<<256, 256, 0, stream>>>(feats, smean, sistd, accum);
  k_prep<<<256, 256, 0, stream>>>(feats, lens, masks, smean, sistd, P, lat, mrow, accum);
  k_norms<<<(G_ * C_) / 256, 256, 0, stream>>>(emb, cn);
  if (packed) k_pack<<<(G_ * 16 * 2 * C_) / 256, 256, 0, stream>>>(W, PB);

  dim3 gq(NROW / 32, G_);
  k_quant<<<gq, 256, 0, stream>>>(lat, emb, cn, target);

  dim3 gl(NROW / 32, G_);
  if (packed)
    k_logits_t<true><<<gl, 256, 0, stream>>>(enc, W, PB, bias, target, mrow, accum, bitmap);
  else
    k_logits_t<false><<<gl, 256, 0, stream>>>(enc, W, PB, bias, target, mrow, accum, bitmap);

  k_final<<<1, 256, 0, stream>>>(accum, bitmap, out);
}
